// GMNAggregator_5583457485045
// MI455X (gfx1250) — matrix operand order repaired and checked by reading, not compiled, not run on hardware
//
#include <hip/hip_runtime.h>
#include <hip/hip_bf16.h>
#include <hip/hip_fp16.h>

typedef __attribute__((ext_vector_type(16))) _Float16 v16h;
typedef __attribute__((ext_vector_type(8)))  _Float16 v8h;
typedef __attribute__((ext_vector_type(4)))  _Float16 v4h;
typedef __attribute__((ext_vector_type(8)))  float    v8f;
typedef __attribute__((ext_vector_type(4)))  float    v4f;
typedef __attribute__((ext_vector_type(4)))  unsigned v4u;

#define D      256          // feature dim (in == out for every layer)
#define TM     128          // rows per block tile
#define PAD    8            // halves of padding per LDS row (breaks 512B bank stride)
#define LDSROW (D + PAD)    // 264 halves = 528B per row, 16B aligned
#define KB8    (D / 32)     // 8 K-steps of 32
#define NG     1024         // number of graphs / segments
#define NTHR   512          // 16 waves per block

// LDS layout (halves):
//   sA : [TM][LDSROW]   activation tile (x or G), f16          66 KB
//   sH : [TM][LDSROW]   hidden tile, f16                        66 KB
//   sB : [D ][LDSROW]   current weight matrix, f16 col-major   132 KB
//   sBatch : int[TM]    batch ids (node kernel only)
#define SA_OFF  0
#define SH_OFF  (TM * LDSROW)
#define SB_OFF  (2 * TM * LDSROW)
#define LDS_HALVES (2 * TM * LDSROW + D * LDSROW)

// ---------------------------------------------------------------------------
// Wave-level GEMM: C[64 x 32] tile of A[128 x 256] * W[256 x 256]
// Fragment layouts per CDNA5 ISA 7.12.2 (f16, 16x16x32):
//   A 16x32 : lane L holds row M=L&15; halves 0..7 -> K=base..base+7,
//             halves 8..15 -> K=base+16..; base = 8*(L>>4).
//   B 32x16 : lane L holds col N=L&15; same K order as A (8*(L>>4)+0..7, +16..
//             (weights staged col-major: sB[n*LDSROW + k])
//   C 16x16 : lane L col N=L&15; vgpr r -> row M = r + 8*(L>>4).
// ---------------------------------------------------------------------------
__device__ __forceinline__ void wave_gemm64x32(const _Float16* __restrict__ sIn,
                                               const _Float16* __restrict__ sB,
                                               int M0, int N0, int lane,
                                               v8f acc[4][2]) {
  const int mr = lane & 15;
  const int hl = lane >> 4;
  v8f z = {};
#pragma unroll
  for (int im = 0; im < 4; ++im)
#pragma unroll
    for (int in = 0; in < 2; ++in) acc[im][in] = z;

  for (int kb = 0; kb < KB8; ++kb) {
    v16h a[4];
#pragma unroll
    for (int im = 0; im < 4; ++im) {
      const _Float16* ap = sIn + (M0 + im * 16 + mr) * LDSROW + kb * 32 + hl * 8;
      v8h lo = *(const v8h*)(ap);
      v8h hi = *(const v8h*)(ap + 16);
      a[im] = __builtin_shufflevector(lo, hi, 0, 1, 2, 3, 4, 5, 6, 7,
                                      8, 9, 10, 11, 12, 13, 14, 15);
    }
    v16h b[2];
#pragma unroll
    for (int in = 0; in < 2; ++in) {
      const _Float16* bp = sB + (N0 + in * 16 + mr) * LDSROW + kb * 32 + hl * 8;
      v8h lo = *(const v8h*)(bp);
      v8h hi = *(const v8h*)(bp + 16);
      b[in] = __builtin_shufflevector(lo, hi, 0, 1, 2, 3, 4, 5, 6, 7,
                                      8, 9, 10, 11, 12, 13, 14, 15);
    }
#pragma unroll
    for (int im = 0; im < 4; ++im)
#pragma unroll
      for (int in = 0; in < 2; ++in)
        acc[im][in] = __builtin_amdgcn_wmma_f32_16x16x32_f16(
            false, a[im], false, b[in], (short)0, acc[im][in], false, false);
  }
}

// Cooperative weight staging: Wt (f16, col-major [D][D]) -> sB [D][LDSROW]
__device__ __forceinline__ void stage_weights(const _Float16* __restrict__ Wt,
                                              _Float16* __restrict__ sB, int t) {
  const int chunk = t & 31;   // 32 x 16B chunks per 512B row
  const int r0 = t >> 5;      // 16 row groups
#pragma unroll
  for (int i = 0; i < 16; ++i) {
    const int row = r0 + 16 * i;
    *(v4u*)(sB + row * LDSROW + chunk * 8) =
        *(const v4u*)(Wt + (size_t)row * D + chunk * 8);
  }
}

// bias + relu epilogue -> f16 LDS tile (wave's own 64x32 strip)
__device__ __forceinline__ void epilogue_relu_lds(const v8f acc[4][2],
                                                  const float* __restrict__ bias,
                                                  _Float16* __restrict__ dst,
                                                  int M0, int N0, int lane) {
  const int mr = lane & 15;
  const int hl = lane >> 4;
#pragma unroll
  for (int in = 0; in < 2; ++in) {
    const int col = N0 + in * 16 + mr;
    const float bv = bias[col];
#pragma unroll
    for (int im = 0; im < 4; ++im) {
#pragma unroll
      for (int r = 0; r < 8; ++r) {
        float v = acc[im][in][r] + bv;
        v = v > 0.f ? v : 0.f;
        dst[(M0 + im * 16 + 8 * hl + r) * LDSROW + col] = (_Float16)v;
      }
    }
  }
}

// ---------------------------------------------------------------------------
// Fused node kernel: both MLPs + sigmoid gate + hadamard + segment-sum
// ---------------------------------------------------------------------------
__global__ void __launch_bounds__(NTHR, 1)
node_kernel(const float* __restrict__ x, const int* __restrict__ batch,
            const _Float16* __restrict__ Wtw1, const float* __restrict__ bw1,
            const _Float16* __restrict__ Wtw2, const float* __restrict__ bw2,
            const _Float16* __restrict__ Wtg1, const float* __restrict__ bg1,
            const _Float16* __restrict__ Wtg2, const float* __restrict__ bg2,
            float* __restrict__ gacc, int nrows) {
  extern __shared__ _Float16 lds[];
  _Float16* sA = lds + SA_OFF;
  _Float16* sH = lds + SH_OFF;
  _Float16* sB = lds + SB_OFF;
  int* sBatch = (int*)(lds + LDS_HALVES);

  const int t = threadIdx.x;
  const int lane = t & 31;
  const int wave = t >> 5;            // 0..15
  const int M0 = (wave >> 3) * 64;    // 2 row groups of 64
  const int N0 = (wave & 7) * 32;     // 8 col groups of 32
  const int mr = lane & 15;
  const int hl = lane >> 4;
  const int row0 = blockIdx.x * TM;

  // ---- stage x tile (f32 -> f16, float4 coalesced) + batch ids + Ww1 ----
  {
    const int c4 = (t & 63) * 4;      // 64 threads per row, 4 floats each
    const int r0 = t >> 6;            // 8 row groups
#pragma unroll
    for (int i = 0; i < 16; ++i) {
      const int row = r0 + 8 * i;
      const int gr = row0 + row;
      v4f v = {};
      if (gr < nrows) v = *(const v4f*)(x + (size_t)gr * D + c4);
      v4h h;
      h.x = (_Float16)v.x; h.y = (_Float16)v.y;
      h.z = (_Float16)v.z; h.w = (_Float16)v.w;
      *(v4h*)(sA + row * LDSROW + c4) = h;
    }
    if (t < TM) {
      const int gr = row0 + t;
      sBatch[t] = (gr < nrows) ? batch[gr] : -1;
    }
    stage_weights(Wtw1, sB, t);
  }
  __syncthreads();

  v8f acc[4][2];
  v8h wpack[4][2];   // 'weighted' tile kept in registers across gate phases

  // ---- phase 1: h = relu(x @ Ww1 + bw1) ----
  wave_gemm64x32(sA, sB, M0, N0, lane, acc);
  epilogue_relu_lds(acc, bw1, sH, M0, N0, lane);
  __syncthreads();
  stage_weights(Wtw2, sB, t);
  __syncthreads();

  // ---- phase 2: weighted = relu(h @ Ww2 + bw2), kept in registers ----
  wave_gemm64x32(sH, sB, M0, N0, lane, acc);
#pragma unroll
  for (int in = 0; in < 2; ++in) {
    const float bv = bw2[N0 + in * 16 + mr];
#pragma unroll
    for (int im = 0; im < 4; ++im)
#pragma unroll
      for (int r = 0; r < 8; ++r) {
        float v = acc[im][in][r] + bv;
        wpack[im][in][r] = (_Float16)(v > 0.f ? v : 0.f);
      }
  }
  __syncthreads();
  stage_weights(Wtg1, sB, t);
  __syncthreads();

  // ---- phase 3: h = relu(x @ Wg1 + bg1) ----
  wave_gemm64x32(sA, sB, M0, N0, lane, acc);
  epilogue_relu_lds(acc, bg1, sH, M0, N0, lane);
  __syncthreads();
  stage_weights(Wtg2, sB, t);
  __syncthreads();

  // ---- phase 4: gate = sigmoid(relu(h @ Wg2 + bg2)); hadamard; segment sum ----
  wave_gemm64x32(sH, sB, M0, N0, lane, acc);
#pragma unroll
  for (int in = 0; in < 2; ++in) {
    const int col = N0 + in * 16 + mr;
    const float bv = bg2[col];
#pragma unroll
    for (int im = 0; im < 4; ++im) {
      const int mbase = M0 + im * 16 + 8 * hl;   // 8 consecutive sorted rows
      int curg = sBatch[mbase];
      float sum = 0.f;
#pragma unroll
      for (int r = 0; r < 8; ++r) {
        const int g = sBatch[mbase + r];
        float zv = acc[im][in][r] + bv;
        zv = zv > 0.f ? zv : 0.f;                       // relu (last MLP layer)
        const float gate = 1.f / (1.f + __expf(-zv));   // sigmoid
        const float val = gate * (float)wpack[im][in][r];
        if (g != curg) {
          if (curg >= 0) atomicAdd(&gacc[(size_t)curg * D + col], sum);
          sum = 0.f;
          curg = g;
        }
        sum += val;
      }
      if (curg >= 0) atomicAdd(&gacc[(size_t)curg * D + col], sum);
    }
  }
}

// ---------------------------------------------------------------------------
// Graph MLP: out = relu(relu(G@Wm1+bm1)@Wm2+bm2), G = [1024 x 256]
// ---------------------------------------------------------------------------
__global__ void __launch_bounds__(NTHR, 1)
graph_kernel(const float* __restrict__ gin,
             const _Float16* __restrict__ Wtm1, const float* __restrict__ bm1,
             const _Float16* __restrict__ Wtm2, const float* __restrict__ bm2,
             float* __restrict__ out) {
  extern __shared__ _Float16 lds[];
  _Float16* sA = lds + SA_OFF;
  _Float16* sH = lds + SH_OFF;
  _Float16* sB = lds + SB_OFF;

  const int t = threadIdx.x;
  const int lane = t & 31;
  const int wave = t >> 5;
  const int M0 = (wave >> 3) * 64;
  const int N0 = (wave & 7) * 32;
  const int mr = lane & 15;
  const int hl = lane >> 4;
  const int row0 = blockIdx.x * TM;

  {
    const int c4 = (t & 63) * 4;
    const int r0 = t >> 6;
#pragma unroll
    for (int i = 0; i < 16; ++i) {
      const int row = r0 + 8 * i;
      v4f v = *(const v4f*)(gin + (size_t)(row0 + row) * D + c4);
      v4h h;
      h.x = (_Float16)v.x; h.y = (_Float16)v.y;
      h.z = (_Float16)v.z; h.w = (_Float16)v.w;
      *(v4h*)(sA + row * LDSROW + c4) = h;
    }
    stage_weights(Wtm1, sB, t);
  }
  __syncthreads();

  v8f acc[4][2];
  wave_gemm64x32(sA, sB, M0, N0, lane, acc);
  epilogue_relu_lds(acc, bm1, sH, M0, N0, lane);
  __syncthreads();
  stage_weights(Wtm2, sB, t);
  __syncthreads();

  wave_gemm64x32(sH, sB, M0, N0, lane, acc);
#pragma unroll
  for (int in = 0; in < 2; ++in) {
    const int col = N0 + in * 16 + mr;
    const float bv = bm2[col];
#pragma unroll
    for (int im = 0; im < 4; ++im)
#pragma unroll
      for (int r = 0; r < 8; ++r) {
        float v = acc[im][in][r] + bv;
        v = v > 0.f ? v : 0.f;
        out[(size_t)(row0 + M0 + im * 16 + 8 * hl + r) * D + col] = v;
      }
  }
}

// Prep: transpose + f32->f16 convert one 256x256 weight (dst[n*D+k] = src[k*D+n])
__global__ void transpose_f16_kernel(const float* __restrict__ src,
                                     _Float16* __restrict__ dst) {
  const int n = blockIdx.x;
  const int k = threadIdx.x;
  dst[(size_t)n * D + k] = (_Float16)src[(size_t)k * D + n];
}

__global__ void zero_f32_kernel(float* __restrict__ p) {
  p[(size_t)blockIdx.x * 256 + threadIdx.x] = 0.f;
}

extern "C" void kernel_launch(void* const* d_in, const int* in_sizes, int n_in,
                              void* d_out, int out_size, void* d_ws, size_t ws_size,
                              hipStream_t stream) {
  const float* x   = (const float*)d_in[0];
  const int* batch = (const int*)d_in[1];
  const float* Ww1 = (const float*)d_in[2];
  const float* bw1 = (const float*)d_in[3];
  const float* Ww2 = (const float*)d_in[4];
  const float* bw2 = (const float*)d_in[5];
  const float* Wg1 = (const float*)d_in[6];
  const float* bg1 = (const float*)d_in[7];
  const float* Wg2 = (const float*)d_in[8];
  const float* bg2 = (const float*)d_in[9];
  const float* Wm1 = (const float*)d_in[10];
  const float* bm1 = (const float*)d_in[11];
  const float* Wm2 = (const float*)d_in[12];
  const float* bm2 = (const float*)d_in[13];

  char* ws = (char*)d_ws;
  const size_t WBYTES = (size_t)D * D * sizeof(_Float16); // 128 KB per matrix
  _Float16* Wt[6];
  for (int i = 0; i < 6; ++i) Wt[i] = (_Float16*)(ws + (size_t)i * WBYTES);
  float* gacc = (float*)(ws + 6 * WBYTES); // [1024 x 256] f32 segment sums

  // 1) weight prep (tiny, stays L2-resident)
  const float* srcs[6] = {Ww1, Ww2, Wg1, Wg2, Wm1, Wm2};
  for (int i = 0; i < 6; ++i)
    transpose_f16_kernel<<<D, 256, 0, stream>>>(srcs[i], Wt[i]);
  zero_f32_kernel<<<(NG * D) / 256, 256, 0, stream>>>(gacc);

  // 2) fused node MLPs + gate + hadamard + segment sum
  const int nrows = in_sizes[0] / D; // 200000
  const int nblk = (nrows + TM - 1) / TM;
  const size_t node_lds = (size_t)LDS_HALVES * sizeof(_Float16) + TM * sizeof(int);
  node_kernel<<<nblk, NTHR, node_lds, stream>>>(x, batch,
                                                Wt[0], bw1, Wt[1], bw2,
                                                Wt[2], bg1, Wt[3], bg2,
                                                gacc, nrows);

  // 3) graph MLP
  const size_t graph_lds = (size_t)LDS_HALVES * sizeof(_Float16);
  graph_kernel<<<NG / TM, NTHR, graph_lds, stream>>>(gacc, Wt[4], bm1, Wt[5], bm2,
                                                     (float*)d_out);
}
